// Tagger_39908836115114
// MI455X (gfx1250) — compile-verified, matrix operand order checked by reading, not run on hardware
//
#include <hip/hip_runtime.h>

// ---------------------------------------------------------------------------
// Decomposable attention (Parikh et al.) forward pass for MI455X (gfx1250).
// All GEMMs run on v_wmma_f32_16x16x32_bf16. wave32 fragment layouts per
// CDNA5 ISA. Feature dims padded: 200->224, 300->320, 400->448.
// ---------------------------------------------------------------------------

typedef __bf16 bf16;
typedef __attribute__((ext_vector_type(16))) __bf16 v16bf;
typedef __attribute__((ext_vector_type(8)))  float  v8f;

#define NP    224           // padded 200
#define NC    448           // padded 400
#define KE    320           // padded 300
#define LSEQ  128
#define BATCH 512
#define ROWS  (BATCH * LSEQ)  // 65536 tokens per side
#define TSTR  144           // LDS transpose-buffer row stride (bf16 elems):
                            // 288 B rows -> 16B-aligned b128 frag loads

// ---- WMMA fragment helpers (wave32) ---------------------------------------
// A-fragment (16x32 bf16, M striped over lanes) and B-fragment (32x16 bf16,
// N striped over lanes) share the same per-lane layout when the B source is
// stored [N][K] row-major. Lane l<16 holds row l, K {kb..kb+7, kb+16..kb+23};
// lane l+16 holds row l, K {kb+8..kb+15, kb+24..kb+31}.
__device__ __forceinline__ v16bf frag_rowmajor(const bf16* base, int stride,
                                               int kbase, int lane) {
  int r = lane & 15, h = (lane >> 4) & 1;
  const bf16* p = base + (size_t)r * stride + kbase + h * 8;
  v16bf f;
#pragma unroll
  for (int i = 0; i < 8; ++i) { f[i] = p[i]; f[8 + i] = p[16 + i]; }
  return f;
}

__device__ __forceinline__ v8f wmma_bf16(v16bf a, v16bf b, v8f c) {
  return __builtin_amdgcn_wmma_f32_16x16x32_bf16(false, a, false, b,
                                                 (short)0, c, false, false);
}

// ---- weight prep: fp32 [Kin][200] -> bf16 [224][Kpad] transposed+padded ----
// mode 0: k valid iff k < R (R = original Kin)
// mode 1: split-concat layout (R=400): k%224<200 maps to (k/224)*200 + k%224
__global__ void prep_weight_kernel(const float* __restrict__ W,
                                   bf16* __restrict__ out,
                                   int R, int Kpad, int mode) {
  int idx = blockIdx.x * blockDim.x + threadIdx.x;
  if (idx >= NP * Kpad) return;
  int o = idx / Kpad, k = idx % Kpad;
  float v = 0.f;
  if (o < 200) {
    if (mode == 0) {
      if (k < R) v = W[(size_t)k * 200 + o];
    } else {
      int seg = k / NP, kk = k % NP;
      if (kk < 200 && seg * 200 + kk < R) v = W[(size_t)(seg * 200 + kk) * 200 + o];
    }
  }
  out[idx] = (bf16)v;
}

__global__ void zero_kernel(float* __restrict__ p, int n) {
  int i = blockIdx.x * blockDim.x + threadIdx.x;
  if (i < n) p[i] = 0.f;
}

// ---- embed + project: out[row][0..223] = emb[tok] @ Wp + bp (bf16) ---------
// One wave per 16-token tile; A gathered+converted from fp32 emb rows.
__global__ void embed_project_kernel(const int* __restrict__ tokens,
                                     const float* __restrict__ emb,
                                     const bf16* __restrict__ WpT,  // [NP][KE]
                                     const float* __restrict__ bp,  // [200]
                                     bf16* __restrict__ out) {      // rows x NC
  int tile = blockIdx.x;
  int lane = threadIdx.x;
  int m = lane & 15, h = (lane >> 4) & 1;
  int row0 = tile * 16;
  int tok = tokens[row0 + m];
  const float* erow = emb + (size_t)tok * 300;

  v16bf af[KE / 32];
#pragma unroll
  for (int c = 0; c < KE / 32; ++c) {
    int kb = c * 32 + h * 8;
#pragma unroll
    for (int i = 0; i < 8; ++i) {
      int k0 = kb + i, k1 = kb + 16 + i;
      af[c][i]     = (bf16)(k0 < 300 ? erow[k0] : 0.f);
      af[c][8 + i] = (bf16)(k1 < 300 ? erow[k1] : 0.f);
    }
  }
  for (int nt = 0; nt < NP / 16; ++nt) {
    v8f acc = {};
    const bf16* wrow = WpT + (size_t)(nt * 16) * KE;
#pragma unroll
    for (int c = 0; c < KE / 32; ++c)
      acc = wmma_bf16(af[c], frag_rowmajor(wrow, KE, c * 32, lane), acc);
    int n = nt * 16 + (lane & 15);
    float bias = (n < 200) ? bp[n] : 0.f;
#pragma unroll
    for (int r = 0; r < 8; ++r)
      out[(size_t)(row0 + r + 8 * h) * NC + n] = (bf16)(acc[r] + bias);
  }
}

// ---- fused 2-layer FF: relu(relu(X@W1+b1)@W2+b2) ---------------------------
// MODE 0: store bf16 Y[row][ys]; MODE 1: ReLU + per-batch sum -> atomicAdd V.
template <int KIN, int MODE>
__global__ void ff2_kernel(const bf16* __restrict__ X, int xs,
                           const bf16* __restrict__ W1T, const float* __restrict__ b1,
                           const bf16* __restrict__ W2T, const float* __restrict__ b2,
                           bf16* __restrict__ Y, int ys,
                           float* __restrict__ V) {
  __shared__ bf16 hid[16 * NP];
  int tile = blockIdx.x;
  int lane = threadIdx.x;
  int h = (lane >> 4) & 1;
  int row0 = tile * 16;
  const bf16* xrow = X + (size_t)row0 * xs;

  v16bf af[KIN / 32];
#pragma unroll
  for (int c = 0; c < KIN / 32; ++c) af[c] = frag_rowmajor(xrow, xs, c * 32, lane);

  for (int nt = 0; nt < NP / 16; ++nt) {
    v8f acc = {};
    const bf16* wrow = W1T + (size_t)(nt * 16) * KIN;
#pragma unroll
    for (int c = 0; c < KIN / 32; ++c)
      acc = wmma_bf16(af[c], frag_rowmajor(wrow, KIN, c * 32, lane), acc);
    int n = nt * 16 + (lane & 15);
    float bias = (n < 200) ? b1[n] : 0.f;
#pragma unroll
    for (int r = 0; r < 8; ++r) {
      float v = acc[r] + bias;
      hid[(r + 8 * h) * NP + n] = (bf16)(v > 0.f ? v : 0.f);
    }
  }
  __syncthreads();

  v16bf hf[NP / 32];
#pragma unroll
  for (int c = 0; c < NP / 32; ++c) hf[c] = frag_rowmajor(hid, NP, c * 32, lane);

  for (int nt = 0; nt < NP / 16; ++nt) {
    v8f acc = {};
    const bf16* wrow = W2T + (size_t)(nt * 16) * NP;
#pragma unroll
    for (int c = 0; c < NP / 32; ++c)
      acc = wmma_bf16(hf[c], frag_rowmajor(wrow, NP, c * 32, lane), acc);
    int n = nt * 16 + (lane & 15);
    float bias = (n < 200) ? b2[n] : 0.f;
    if (MODE == 0) {
#pragma unroll
      for (int r = 0; r < 8; ++r) {
        float v = acc[r] + bias;
        Y[(size_t)(row0 + r + 8 * h) * ys + n] = (bf16)(v > 0.f ? v : 0.f);
      }
    } else {
      float s = 0.f;
#pragma unroll
      for (int r = 0; r < 8; ++r) {
        float v = acc[r] + bias;
        s += (v > 0.f ? v : 0.f);
      }
      atomicAdd(&V[(size_t)(row0 / LSEQ) * NP + n], s);
    }
  }
}

// ---- attention: E = fa.fb^T, dual softmax, beta/alpha into concat halves ---
// One 256-thread (8-wave) block per batch element.
// Dynamic LDS (128 KB): Ef fp32 128x128 (64KB, later reused as a bf16
// transpose buffer Tb[224][TSTR]) + Ps bf16 128x128 + Qt bf16 128x128.
__global__ void attn_kernel(const bf16* __restrict__ fa, const bf16* __restrict__ fb,
                            bf16* __restrict__ wa, bf16* __restrict__ wb) {
  extern __shared__ char smem[];
  float* Ef = (float*)smem;
  bf16* Tb = (bf16*)smem;                    // overlays Ef once softmax done
  bf16* Ps = (bf16*)(smem + LSEQ * LSEQ * 4);
  bf16* Qt = Ps + LSEQ * LSEQ;
  int bi = blockIdx.x;
  int tid = threadIdx.x;
  int w = tid >> 5, lane = tid & 31;
  int h = (lane >> 4) & 1;
  const bf16* fap = fa + (size_t)bi * LSEQ * NP;
  const bf16* fbp = fb + (size_t)bi * LSEQ * NP;

  // E tiles: wave w owns m-tile w, loops all 8 h-tiles. fb rows serve as
  // [N][K]-stored B directly (E = fa . fb^T).
  v16bf af[NP / 32];
#pragma unroll
  for (int c = 0; c < NP / 32; ++c)
    af[c] = frag_rowmajor(fap + (size_t)(w * 16) * NP, NP, c * 32, lane);
  for (int ht = 0; ht < 8; ++ht) {
    v8f acc = {};
#pragma unroll
    for (int c = 0; c < NP / 32; ++c)
      acc = wmma_bf16(af[c], frag_rowmajor(fbp + (size_t)(ht * 16) * NP, NP, c * 32, lane), acc);
    int n = ht * 16 + (lane & 15);
#pragma unroll
    for (int r = 0; r < 8; ++r)
      Ef[(w * 16 + r + 8 * h) * LSEQ + n] = acc[r];
  }
  __syncthreads();

  if (tid < LSEQ) {                 // row softmax -> Ps[p][h]
    int p = tid;
    float mx = -1e30f;
    for (int j = 0; j < LSEQ; ++j) mx = fmaxf(mx, Ef[p * LSEQ + j]);
    float s = 0.f;
    for (int j = 0; j < LSEQ; ++j) s += __expf(Ef[p * LSEQ + j] - mx);
    float inv = 1.f / s;
    for (int j = 0; j < LSEQ; ++j)
      Ps[p * LSEQ + j] = (bf16)(__expf(Ef[p * LSEQ + j] - mx) * inv);
  } else {                          // column softmax -> Qt[h][p] (transposed)
    int hc = tid - LSEQ;
    float mx = -1e30f;
    for (int j = 0; j < LSEQ; ++j) mx = fmaxf(mx, Ef[j * LSEQ + hc]);
    float s = 0.f;
    for (int j = 0; j < LSEQ; ++j) s += __expf(Ef[j * LSEQ + hc] - mx);
    float inv = 1.f / s;
    for (int j = 0; j < LSEQ; ++j)
      Qt[hc * LSEQ + j] = (bf16)(__expf(Ef[j * LSEQ + hc] - mx) * inv);
  }
  __syncthreads();   // Ef dead from here; Tb overlay becomes usable

  const bf16* arow = wa + (size_t)bi * LSEQ * NC;  // a in cols [0,224)
  const bf16* brow = wb + (size_t)bi * LSEQ * NC;  // b in cols [0,224)

  // ---- stage b^T into LDS: Tb[d][h] = b[h][d] (coalesced global reads) ----
  for (int idx = tid; idx < LSEQ * NP; idx += 256) {
    int hh = idx / NP, d = idx - hh * NP;
    Tb[d * TSTR + hh] = brow[(size_t)hh * NC + d];
  }
  __syncthreads();

  // beta = Ps @ b  -> wa cols [224,448)
  v16bf pf[LSEQ / 32];
#pragma unroll
  for (int c = 0; c < LSEQ / 32; ++c)
    pf[c] = frag_rowmajor(Ps + (w * 16) * LSEQ, LSEQ, c * 32, lane);
  for (int nt = 0; nt < NP / 16; ++nt) {
    v8f acc = {};
#pragma unroll
    for (int c = 0; c < LSEQ / 32; ++c)
      acc = wmma_bf16(pf[c], frag_rowmajor(Tb + (nt * 16) * TSTR, TSTR, c * 32, lane), acc);
    int n = nt * 16 + (lane & 15);
#pragma unroll
    for (int r = 0; r < 8; ++r)
      wa[((size_t)bi * LSEQ + w * 16 + r + 8 * h) * NC + NP + n] = (bf16)acc[r];
  }
  __syncthreads();

  // ---- stage a^T into LDS: Tb[d][p] = a[p][d] ----
  for (int idx = tid; idx < LSEQ * NP; idx += 256) {
    int hh = idx / NP, d = idx - hh * NP;
    Tb[d * TSTR + hh] = arow[(size_t)hh * NC + d];
  }
  __syncthreads();

  // alpha = Qt @ a -> wb cols [224,448)
  v16bf qf[LSEQ / 32];
#pragma unroll
  for (int c = 0; c < LSEQ / 32; ++c)
    qf[c] = frag_rowmajor(Qt + (w * 16) * LSEQ, LSEQ, c * 32, lane);
  for (int nt = 0; nt < NP / 16; ++nt) {
    v8f acc = {};
#pragma unroll
    for (int c = 0; c < LSEQ / 32; ++c)
      acc = wmma_bf16(qf[c], frag_rowmajor(Tb + (nt * 16) * TSTR, TSTR, c * 32, lane), acc);
    int n = nt * 16 + (lane & 15);
#pragma unroll
    for (int r = 0; r < 8; ++r)
      wb[((size_t)bi * LSEQ + w * 16 + r + 8 * h) * NC + NP + n] = (bf16)acc[r];
  }
}

// ---- head: h = ff(concat(v1,v2)); out = h @ Ow + Ob ------------------------
__global__ void head_kernel(const float* __restrict__ v1, const float* __restrict__ v2,
                            const bf16* __restrict__ H1T, const float* __restrict__ h1b,
                            const bf16* __restrict__ H2T, const float* __restrict__ h2b,
                            const float* __restrict__ Ow, const float* __restrict__ Ob,
                            float* __restrict__ out) {
  __shared__ bf16 Xs[16 * NC];
  __shared__ bf16 Hs[16 * NP];
  int lane = threadIdx.x;
  int h = (lane >> 4) & 1;
  int b0 = blockIdx.x * 16;

  for (int idx = lane; idx < 16 * NC; idx += 32) {
    int m = idx / NC, c = idx % NC;
    float v = (c < NP) ? v1[(size_t)(b0 + m) * NP + c]
                       : v2[(size_t)(b0 + m) * NP + (c - NP)];
    Xs[idx] = (bf16)v;
  }
  __syncthreads();

  v16bf af[NC / 32];
#pragma unroll
  for (int c = 0; c < NC / 32; ++c) af[c] = frag_rowmajor(Xs, NC, c * 32, lane);
  for (int nt = 0; nt < NP / 16; ++nt) {
    v8f acc = {};
    const bf16* wrow = H1T + (size_t)(nt * 16) * NC;
#pragma unroll
    for (int c = 0; c < NC / 32; ++c)
      acc = wmma_bf16(af[c], frag_rowmajor(wrow, NC, c * 32, lane), acc);
    int n = nt * 16 + (lane & 15);
    float bias = (n < 200) ? h1b[n] : 0.f;
#pragma unroll
    for (int r = 0; r < 8; ++r) {
      float v = acc[r] + bias;
      Hs[(r + 8 * h) * NP + n] = (bf16)(v > 0.f ? v : 0.f);
    }
  }
  __syncthreads();

  v16bf hf[NP / 32];
#pragma unroll
  for (int c = 0; c < NP / 32; ++c) hf[c] = frag_rowmajor(Hs, NP, c * 32, lane);
  __syncthreads();
  for (int nt = 0; nt < NP / 16; ++nt) {
    v8f acc = {};
    const bf16* wrow = H2T + (size_t)(nt * 16) * NP;
#pragma unroll
    for (int c = 0; c < NP / 32; ++c)
      acc = wmma_bf16(hf[c], frag_rowmajor(wrow, NP, c * 32, lane), acc);
    int n = nt * 16 + (lane & 15);
    float bias = (n < 200) ? h2b[n] : 0.f;
#pragma unroll
    for (int r = 0; r < 8; ++r) {
      float v = acc[r] + bias;
      Hs[(r + 8 * h) * NP + n] = (bf16)(v > 0.f ? v : 0.f);
    }
  }
  __syncthreads();

  if (lane < 16) {
    int m = lane;
    for (int t = 0; t < 3; ++t) {
      float s = Ob[t];
      for (int k = 0; k < 200; ++k) s += (float)Hs[m * NP + k] * Ow[k * 3 + t];
      out[(size_t)(b0 + m) * 3 + t] = s;
    }
  }
}

// ---------------------------------------------------------------------------
extern "C" void kernel_launch(void* const* d_in, const int* in_sizes, int n_in,
                              void* d_out, int out_size, void* d_ws, size_t ws_size,
                              hipStream_t stream) {
  (void)in_sizes; (void)n_in; (void)out_size; (void)ws_size;
  const int*   premise = (const int*)d_in[0];
  const int*   hyp     = (const int*)d_in[1];
  const float* emb = (const float*)d_in[2];
  const float* Wp  = (const float*)d_in[3];
  const float* bp  = (const float*)d_in[4];
  const float* F1w = (const float*)d_in[5];  const float* F1b = (const float*)d_in[6];
  const float* F2w = (const float*)d_in[7];  const float* F2b = (const float*)d_in[8];
  const float* G1w = (const float*)d_in[9];  const float* G1b = (const float*)d_in[10];
  const float* G2w = (const float*)d_in[11]; const float* G2b = (const float*)d_in[12];
  const float* H1w = (const float*)d_in[13]; const float* H1b = (const float*)d_in[14];
  const float* H2w = (const float*)d_in[15]; const float* H2b = (const float*)d_in[16];
  const float* Ow  = (const float*)d_in[17]; const float* Ob  = (const float*)d_in[18];
  float* out = (float*)d_out;

  char* ws = (char*)d_ws;
  size_t off = 0;
  auto alloc = [&](size_t bytes) {
    off = (off + 255) & ~(size_t)255;
    char* p = ws + off;
    off += bytes;
    return p;
  };
  bf16* WpT = (bf16*)alloc((size_t)NP * KE * 2);
  bf16* F1T = (bf16*)alloc((size_t)NP * NP * 2);
  bf16* F2T = (bf16*)alloc((size_t)NP * NP * 2);
  bf16* G1T = (bf16*)alloc((size_t)NP * NC * 2);
  bf16* G2T = (bf16*)alloc((size_t)NP * NP * 2);
  bf16* H1T = (bf16*)alloc((size_t)NP * NC * 2);
  bf16* H2T = (bf16*)alloc((size_t)NP * NP * 2);
  bf16* wa  = (bf16*)alloc((size_t)ROWS * NC * 2);
  bf16* wb  = (bf16*)alloc((size_t)ROWS * NC * 2);
  bf16* fa  = (bf16*)alloc((size_t)ROWS * NP * 2);
  bf16* fb  = (bf16*)alloc((size_t)ROWS * NP * 2);
  float* v1 = (float*)alloc((size_t)BATCH * NP * 4);
  float* v2 = (float*)alloc((size_t)BATCH * NP * 4);

  const int T = 256;
  prep_weight_kernel<<<(NP * KE + T - 1) / T, T, 0, stream>>>(Wp,  WpT, 300, KE, 0);
  prep_weight_kernel<<<(NP * NP + T - 1) / T, T, 0, stream>>>(F1w, F1T, 200, NP, 0);
  prep_weight_kernel<<<(NP * NP + T - 1) / T, T, 0, stream>>>(F2w, F2T, 200, NP, 0);
  prep_weight_kernel<<<(NP * NC + T - 1) / T, T, 0, stream>>>(G1w, G1T, 400, NC, 1);
  prep_weight_kernel<<<(NP * NP + T - 1) / T, T, 0, stream>>>(G2w, G2T, 200, NP, 0);
  prep_weight_kernel<<<(NP * NC + T - 1) / T, T, 0, stream>>>(H1w, H1T, 400, NC, 1);
  prep_weight_kernel<<<(NP * NP + T - 1) / T, T, 0, stream>>>(H2w, H2T, 200, NP, 0);
  zero_kernel<<<(BATCH * NP + T - 1) / T, T, 0, stream>>>(v1, BATCH * NP);
  zero_kernel<<<(BATCH * NP + T - 1) / T, T, 0, stream>>>(v2, BATCH * NP);

  embed_project_kernel<<<ROWS / 16, 32, 0, stream>>>(premise, emb, WpT, bp, wa);
  embed_project_kernel<<<ROWS / 16, 32, 0, stream>>>(hyp,     emb, WpT, bp, wb);

  ff2_kernel<NP, 0><<<ROWS / 16, 32, 0, stream>>>(wa, NC, F1T, F1b, F2T, F2b, fa, NP, nullptr);
  ff2_kernel<NP, 0><<<ROWS / 16, 32, 0, stream>>>(wb, NC, F1T, F1b, F2T, F2b, fb, NP, nullptr);

  size_t attn_lds = (size_t)LSEQ * LSEQ * 4 + 2 * (size_t)LSEQ * LSEQ * 2;  // 128 KB
  attn_kernel<<<BATCH, 256, attn_lds, stream>>>(fa, fb, wa, wb);

  ff2_kernel<NC, 1><<<ROWS / 16, 32, 0, stream>>>(wa, NC, G1T, G1b, G2T, G2b, nullptr, 0, v1);
  ff2_kernel<NC, 1><<<ROWS / 16, 32, 0, stream>>>(wb, NC, G1T, G1b, G2T, G2b, nullptr, 0, v2);

  head_kernel<<<BATCH / 16, 32, 0, stream>>>(v1, v2, H1T, H1b, H2T, H2b, Ow, Ob, out);
}
